// GeometricCrossAttention_68710886802023
// MI455X (gfx1250) — matrix operand order repaired and checked by reading, not compiled, not run on hardware
//
#include <hip/hip_runtime.h>
#include <hip/hip_bf16.h>

// ---------------------------------------------------------------------------
// GeometricCrossAttention for MI455X (gfx1250, wave32, WMMA, async-LDS)
//   B=4, L=256, P=1024, D=512, H=8, RBF=50, d=64
// ---------------------------------------------------------------------------

typedef __bf16 bhalf;
typedef __attribute__((ext_vector_type(16))) __bf16 v16bf;
typedef __attribute__((ext_vector_type(8)))  float  v8f;

#define BB   4
#define LL   256
#define PP   1024
#define DD   512
#define HH   8
#define NRBF 50
#define DH   64

// ------------------------- async global->LDS support -----------------------
#if defined(__has_builtin)
#if __has_builtin(__builtin_amdgcn_global_load_async_to_lds_b128) && \
    __has_builtin(__builtin_amdgcn_global_load_async_to_lds_b64) &&  \
    __has_builtin(__builtin_amdgcn_s_wait_asynccnt)
#define ASYNC_LDS 1
#endif
#endif

#if ASYNC_LDS
// builtin prototype (from clang diagnostics):
//   b128: (v4i addrspace(1)*, v4i addrspace(3)*, imm offset, imm cpol)
//   b64 : (v2i addrspace(1)*, v2i addrspace(3)*, imm offset, imm cpol)
typedef __attribute__((__vector_size__(4 * sizeof(int)))) int vi4;
typedef __attribute__((__vector_size__(2 * sizeof(int)))) int vi2;

__device__ inline void async_copy_b128(const float* g, float* l) {
  __builtin_amdgcn_global_load_async_to_lds_b128(
      (__attribute__((address_space(1))) vi4*)g,
      (__attribute__((address_space(3))) vi4*)l, 0, 0);
}
__device__ inline void async_copy_b64(const float* g, float* l) {
  __builtin_amdgcn_global_load_async_to_lds_b64(
      (__attribute__((address_space(1))) vi2*)g,
      (__attribute__((address_space(3))) vi2*)l, 0, 0);
}
#define ASYNC_WAIT() __builtin_amdgcn_s_wait_asynccnt(0)
#endif

// ---------------------------------------------------------------------------
// WMMA fragment loaders (ISA 7.12.2 layouts, wave32)
// A 16x32 bf16: lane m=lane&15; elems 0..7 -> K=kb..kb+7, elems 8..15 -> +16,
//               kb = (lane>=16)?8:0
// B 32x16 bf16: lane n=lane&15; elems 0..7 -> K=kb..kb+7, elems 8..15 -> +16, kb = (lane>=16)?8:0
// C/D 16x16 f32: VGPR r: lanes0-15 M=r, lanes16-31 M=8+r; N=lane&15
// ---------------------------------------------------------------------------

__device__ inline v16bf load_a_frag_bf16(const bhalf* __restrict__ base_row0,
                                         int ld, int k0) {
  const int lane = threadIdx.x & 31;
  const int m  = lane & 15;
  const int kb = (lane >> 4) << 3;
  const bhalf* p = base_row0 + (size_t)m * ld + k0;
  v16bf a;
#pragma unroll
  for (int j = 0; j < 8; ++j) a[j] = p[kb + j];
#pragma unroll
  for (int j = 0; j < 8; ++j) a[8 + j] = p[16 + kb + j];
  return a;
}

// B[k,n] = K[p0+n, kk0+k]  (K^T operand; contiguous in k per lane)
__device__ inline v16bf load_bT_frag_bf16(const bhalf* __restrict__ base,
                                          int ld, int kk0) {
  const int lane = threadIdx.x & 31;
  const int nn = lane & 15;
  const int kb = (lane >> 4) << 3;
  const bhalf* p = base + (size_t)nn * ld + kk0 + kb;
  v16bf b;
#pragma unroll
  for (int j = 0; j < 8; ++j) b[j] = p[j];
#pragma unroll
  for (int j = 0; j < 8; ++j) b[8 + j] = p[16 + j];
  return b;
}

// B[k,n] = V[p0+k, d0+n]  (row-major V operand; strided in k)
__device__ inline v16bf load_b_frag_rm(const bhalf* __restrict__ base, int ld) {
  const int lane = threadIdx.x & 31;
  const int nn = lane & 15;
  const int kb = (lane >> 4) << 3;
  const bhalf* p = base + (size_t)kb * ld + nn;
  v16bf b;
#pragma unroll
  for (int j = 0; j < 8; ++j) b[j] = p[(size_t)j * ld];
#pragma unroll
  for (int j = 0; j < 8; ++j) b[8 + j] = p[(size_t)(16 + j) * ld];
  return b;
}

#define WMMA_BF16(A, Bm, C) \
  __builtin_amdgcn_wmma_f32_16x16x32_bf16(false, (A), false, (Bm), (short)0, (C), false, false)

// ---------------------------------------------------------------------------
// Staged GEMM core: block covers one 16-row tile x 128-col window; wave w owns
// columns [nwin + 16w, nwin + 16w + 16).  A/W tiles staged into padded LDS
// (async global->LDS when available), fragments built from LDS with f32->bf16
// conversion.  K = 512 stepped by 32.
// ---------------------------------------------------------------------------
__device__ inline v8f gemm_k512_staged(const float* __restrict__ A,
                                       const float* __restrict__ W,
                                       int row0, int nwin, int N, int K,
                                       float (*sW)[132], float (*sA)[36]) {
  const int wave = threadIdx.x >> 5;
  const int lane = threadIdx.x & 31;
  const int nn = lane & 15;
  v8f acc = {};
  for (int k0 = 0; k0 < K; k0 += 32) {
    __syncthreads();  // protect LDS tiles from previous iteration
    {  // stage W[k0..k0+32) x [nwin..nwin+128): 256 thr x 16 f32 (coalesced)
      const int kk = threadIdx.x >> 3;
      const int ch = (threadIdx.x & 7) << 4;
      const float* src = W + (size_t)(k0 + kk) * N + nwin + ch;
      float* dst = &sW[kk][ch];
#if ASYNC_LDS
#pragma unroll
      for (int j = 0; j < 4; ++j) async_copy_b128(src + 4 * j, dst + 4 * j);
#else
#pragma unroll
      for (int j = 0; j < 16; ++j) dst[j] = src[j];
#endif
    }
    if (threadIdx.x < 128) {  // stage A[row0..+16) x [k0..k0+32): 128 thr x 4
      const int m = threadIdx.x >> 3;
      const int ch = (threadIdx.x & 7) << 2;
      const float* src = A + (size_t)(row0 + m) * K + k0 + ch;
#if ASYNC_LDS
      async_copy_b128(src, &sA[m][ch]);
#else
#pragma unroll
      for (int j = 0; j < 4; ++j) sA[m][ch + j] = src[j];
#endif
    }
#if ASYNC_LDS
    ASYNC_WAIT();
#endif
    __syncthreads();

    v16bf a, b;
    {
      const int m = lane & 15;
      const int kb = (lane >> 4) << 3;
#pragma unroll
      for (int j = 0; j < 8; ++j) a[j] = (__bf16)sA[m][kb + j];
#pragma unroll
      for (int j = 0; j < 8; ++j) a[8 + j] = (__bf16)sA[m][16 + kb + j];
    }
    {
      const int kb = (lane >> 4) << 3;
#pragma unroll
      for (int j = 0; j < 8; ++j) b[j] = (__bf16)sW[kb + j][wave * 16 + nn];
#pragma unroll
      for (int j = 0; j < 8; ++j) b[8 + j] = (__bf16)sW[16 + kb + j][wave * 16 + nn];
    }
    acc = WMMA_BF16(a, b, acc);
  }
  return acc;
}

// ---------------------------------------------------------------------------
// Kernel 1: projection GEMM   C_bf16[M,512] = bf16( A[M,512] @ W + bias )
// ---------------------------------------------------------------------------
__global__ __launch_bounds__(256) void proj_gemm_bf16(
    const float* __restrict__ A, const float* __restrict__ W,
    const float* __restrict__ bias, bhalf* __restrict__ C, int M) {
  __shared__ float sW[32][132];
  __shared__ float sA[16][36];
  const int bpr = (DD >> 4) >> 3;                 // 128-col windows per row = 4
  const int tm = blockIdx.x / bpr;
  const int nwin = (blockIdx.x % bpr) << 7;
  const int wave = threadIdx.x >> 5;
  const int lane = threadIdx.x & 31;
  v8f acc = gemm_k512_staged(A, W, tm * 16, nwin, DD, DD, sW, sA);
  const int n = nwin + wave * 16 + (lane & 15);
  const float bv = bias[n];
  const int mbase = tm * 16 + ((lane >> 4) << 3);
#pragma unroll
  for (int r = 0; r < 8; ++r)
    C[(size_t)(mbase + r) * DD + n] = (__bf16)(acc[r] + bv);
}

// ---------------------------------------------------------------------------
// Kernel 2: logits[b,h,l,p] = QK^T/8 + (rbf@Wr + br), masked.
// Block = (b, 16-row L-tile, 64-wide P-chunk); wave = head.
// rbf rows staged per-thread via async b64 (rows only 8B aligned).
// ---------------------------------------------------------------------------
__global__ __launch_bounds__(256) void bias_logits(
    const bhalf* __restrict__ Qbf, const bhalf* __restrict__ Kbf,
    const float* __restrict__ rbf, const int* __restrict__ kmask,
    const float* __restrict__ Wr, const float* __restrict__ br,
    float* __restrict__ logits) {
  const int b  = blockIdx.x >> 8;
  const int lt = (blockIdx.x >> 4) & 15;
  const int pc = blockIdx.x & 15;
  const int l0 = lt << 4;
  const int h    = threadIdx.x >> 5;   // wave id == head
  const int lane = threadIdx.x & 31;
  const int n  = lane & 15;
  const int mb = (lane >> 4) << 3;

  __shared__ float sWr[NRBF][HH];
  __shared__ float sbr[HH];
  __shared__ float sRbf[16][16][NRBF];   // one 50-f32 row per thread
  __shared__ float sBias[16][16][HH];

  for (int i = threadIdx.x; i < NRBF * HH; i += 256) sWr[i / HH][i % HH] = Wr[i];
  if (threadIdx.x < HH) sbr[threadIdx.x] = br[threadIdx.x];

  const bhalf* qbase = Qbf + ((size_t)b * LL + l0) * DD + h * DH;
  const v16bf aq0 = load_a_frag_bf16(qbase, DD, 0);
  const v16bf aq1 = load_a_frag_bf16(qbase, DD, 32);

  const int ll = threadIdx.x >> 4;
  const int pp = threadIdx.x & 15;

  for (int t = 0; t < 4; ++t) {
    const int p0 = (pc << 6) + (t << 4);
    __syncthreads();  // sBias reuse + first-iteration sWr visibility
    const float* src = rbf + (((size_t)b * LL + l0 + ll) * PP + p0 + pp) * NRBF;
    float* dstrow = &sRbf[ll][pp][0];
#if ASYNC_LDS
#pragma unroll
    for (int j = 0; j < 25; ++j) async_copy_b64(src + 2 * j, dstrow + 2 * j);
    ASYNC_WAIT();  // thread reads back only its own staged row
#else
    for (int i = 0; i < NRBF; ++i) dstrow[i] = src[i];
#endif
    if (t < 3) __builtin_prefetch(src + 16 * NRBF, 0, 1);  // next p-tile
    float accb[HH] = {0.f, 0.f, 0.f, 0.f, 0.f, 0.f, 0.f, 0.f};
    for (int i = 0; i < NRBF; ++i) {
      const float x = dstrow[i];
#pragma unroll
      for (int hh = 0; hh < HH; ++hh) accb[hh] += x * sWr[i][hh];
    }
#pragma unroll
    for (int hh = 0; hh < HH; ++hh) sBias[ll][pp][hh] = accb[hh] + sbr[hh];
    __syncthreads();

    v8f s = {};
    const bhalf* kb2 = Kbf + ((size_t)b * PP + p0) * DD + h * DH;
    const v16bf bk0 = load_bT_frag_bf16(kb2, DD, 0);
    const v16bf bk1 = load_bT_frag_bf16(kb2, DD, 32);
    s = WMMA_BF16(aq0, bk0, s);
    s = WMMA_BF16(aq1, bk1, s);

    const int mval = kmask[b * PP + p0 + n];
#pragma unroll
    for (int r = 0; r < 8; ++r) {
      float x = s[r] * 0.125f + sBias[mb + r][n][h];  // 1/sqrt(64)
      x = (mval == 0) ? -1e9f : x;
      logits[(((size_t)(b * HH + h)) * LL + l0 + mb + r) * PP + p0 + n] = x;
    }
  }
}

// ---------------------------------------------------------------------------
// Kernel 3: per-row softmax stats over contiguous 1024-float logits rows.
// stats[row] = {rowmax, 1/sumexp}.  One wave per row, 8 rows per block.
// ---------------------------------------------------------------------------
__global__ __launch_bounds__(256) void softmax_stats(
    const float* __restrict__ logits, float* __restrict__ stats) {
  const int wave = threadIdx.x >> 5;
  const int lane = threadIdx.x & 31;
  const int row = blockIdx.x * 8 + wave;          // B*H*L = 8192 rows
  const float* xr = logits + (size_t)row * PP;
  float v[32];
  float mx = -1e30f;
#pragma unroll
  for (int i = 0; i < 32; ++i) {
    v[i] = xr[lane + 32 * i];
    mx = fmaxf(mx, v[i]);
  }
  mx = fmaxf(mx, __shfl_xor(mx, 1));
  mx = fmaxf(mx, __shfl_xor(mx, 2));
  mx = fmaxf(mx, __shfl_xor(mx, 4));
  mx = fmaxf(mx, __shfl_xor(mx, 8));
  mx = fmaxf(mx, __shfl_xor(mx, 16));
  float s = 0.f;
#pragma unroll
  for (int i = 0; i < 32; ++i) s += __expf(v[i] - mx);
  s += __shfl_xor(s, 1);
  s += __shfl_xor(s, 2);
  s += __shfl_xor(s, 4);
  s += __shfl_xor(s, 8);
  s += __shfl_xor(s, 16);
  if (lane == 0) {
    stats[(size_t)row * 2 + 0] = mx;
    stats[(size_t)row * 2 + 1] = 1.0f / s;
  }
}

// ---------------------------------------------------------------------------
// Kernel 4: attn-mean over heads:  mean[b,l,p] = (1/8) sum_h softmax(logit)
// ---------------------------------------------------------------------------
__global__ __launch_bounds__(256) void attn_mean(
    const float* __restrict__ logits, const float* __restrict__ stats,
    float* __restrict__ meanout) {
  const size_t idx = (size_t)blockIdx.x * 256 + threadIdx.x;  // B*L*P
  const int p = idx & (PP - 1);
  const int l = (idx >> 10) & (LL - 1);
  const int b = idx >> 18;
  float acc = 0.f;
#pragma unroll
  for (int h = 0; h < HH; ++h) {
    const size_t row = ((size_t)(b * HH + h)) * LL + l;
    const float m = stats[row * 2 + 0];
    const float r = stats[row * 2 + 1];
    acc += __expf(logits[row * PP + p] - m) * r;
  }
  meanout[idx] = acc * 0.125f;
}

// ---------------------------------------------------------------------------
// Kernel 5: attn @ V.  Block = (b, L-tile, head); wave w owns p in [128w,128w+128).
// Probs (f32) -> bf16 staged through per-wave LDS for C->A layout transpose,
// 4 WMMAs per 32-p chunk; cross-wave reduction through LDS (deterministic).
// ---------------------------------------------------------------------------
__global__ __launch_bounds__(256) void attn_av(
    const bhalf* __restrict__ Vbf, const float* __restrict__ logits,
    const float* __restrict__ stats, float* __restrict__ attnout) {
  const int b  = blockIdx.x >> 7;
  const int lt = (blockIdx.x >> 3) & 15;
  const int h  = blockIdx.x & 7;
  const int l0 = lt << 4;
  const int w    = threadIdx.x >> 5;
  const int lane = threadIdx.x & 31;
  const int n  = lane & 15;
  const int mb = (lane >> 4) << 3;

  __shared__ bhalf sProbs[8][16][32];
  __shared__ float sO[8][16][64];

  const size_t rowbase = ((size_t)(b * HH + h)) * LL + l0;
  float mrow[8], rrow[8];
#pragma unroll
  for (int r = 0; r < 8; ++r) {
    mrow[r] = stats[(rowbase + mb + r) * 2 + 0];
    rrow[r] = stats[(rowbase + mb + r) * 2 + 1];
  }

  v8f o0 = {}, o1 = {}, o2 = {}, o3 = {};
  for (int c = 0; c < 4; ++c) {
    const int p0 = w * 128 + c * 32;
#pragma unroll
    for (int sub = 0; sub < 2; ++sub) {
#pragma unroll
      for (int r = 0; r < 8; ++r) {
        const float x = logits[(rowbase + mb + r) * PP + p0 + sub * 16 + n];
        const float pr = __expf(x - mrow[r]) * rrow[r];
        sProbs[w][mb + r][sub * 16 + n] = (__bf16)pr;  // C->A staging (in-wave)
      }
    }
    v16bf ap;
    {
      const int m = lane & 15;
      const int kb = (lane >> 4) << 3;
#pragma unroll
      for (int j = 0; j < 8; ++j) ap[j] = sProbs[w][m][kb + j];
#pragma unroll
      for (int j = 0; j < 8; ++j) ap[8 + j] = sProbs[w][m][16 + kb + j];
    }
    const bhalf* vbase = Vbf + ((size_t)b * PP + p0) * DD + h * DH;
    const v16bf bv0 = load_b_frag_rm(vbase + 0, DD);
    const v16bf bv1 = load_b_frag_rm(vbase + 16, DD);
    const v16bf bv2 = load_b_frag_rm(vbase + 32, DD);
    const v16bf bv3 = load_b_frag_rm(vbase + 48, DD);
    o0 = WMMA_BF16(ap, bv0, o0);
    o1 = WMMA_BF16(ap, bv1, o1);
    o2 = WMMA_BF16(ap, bv2, o2);
    o3 = WMMA_BF16(ap, bv3, o3);
  }
#pragma unroll
  for (int r = 0; r < 8; ++r) {
    sO[w][mb + r][0 + n]  = o0[r];
    sO[w][mb + r][16 + n] = o1[r];
    sO[w][mb + r][32 + n] = o2[r];
    sO[w][mb + r][48 + n] = o3[r];
  }
  __syncthreads();
  for (int i = threadIdx.x; i < 1024; i += 256) {
    const int ll = i >> 6;
    const int dd = i & 63;
    float a = 0.f;
#pragma unroll
    for (int ww = 0; ww < 8; ++ww) a += sO[ww][ll][dd];  // fixed order: determ.
    attnout[((size_t)b * LL + l0 + ll) * DD + h * DH + dd] = a;
  }
}

// ---------------------------------------------------------------------------
// Kernel 6: out-projection + residual:  x = attnout @ Wo + bo + query
// ---------------------------------------------------------------------------
__global__ __launch_bounds__(256) void outproj_residual(
    const float* __restrict__ Ao, const float* __restrict__ Wo,
    const float* __restrict__ bo, const float* __restrict__ query,
    float* __restrict__ xbuf) {
  __shared__ float sW[32][132];
  __shared__ float sA[16][36];
  const int tm = blockIdx.x >> 2;
  const int nwin = (blockIdx.x & 3) << 7;
  const int wave = threadIdx.x >> 5;
  const int lane = threadIdx.x & 31;
  v8f acc = gemm_k512_staged(Ao, Wo, tm * 16, nwin, DD, DD, sW, sA);
  const int n = nwin + wave * 16 + (lane & 15);
  const float bb = bo[n];
  const int mbase = tm * 16 + ((lane >> 4) << 3);
#pragma unroll
  for (int r = 0; r < 8; ++r) {
    const size_t idx = (size_t)(mbase + r) * DD + n;
    xbuf[idx] = acc[r] + bb + query[idx];
  }
}

// ---------------------------------------------------------------------------
// Kernel 7: layernorm over D=512 per row
// ---------------------------------------------------------------------------
__global__ __launch_bounds__(256) void layernorm_rows(
    const float* __restrict__ x, const float* __restrict__ gamma,
    const float* __restrict__ beta, float* __restrict__ out) {
  __shared__ float s1[256];
  __shared__ float s2[256];
  const int row = blockIdx.x;
  const int t = threadIdx.x;
  const float* xr = x + (size_t)row * DD;
  const float a = xr[t];
  const float c = xr[t + 256];
  s1[t] = a + c;
  s2[t] = a * a + c * c;
  __syncthreads();
  for (int off = 128; off > 0; off >>= 1) {
    if (t < off) { s1[t] += s1[t + off]; s2[t] += s2[t + off]; }
    __syncthreads();
  }
  const float mu  = s1[0] * (1.0f / 512.0f);
  const float var = s2[0] * (1.0f / 512.0f) - mu * mu;
  const float inv = rsqrtf(var + 1e-5f);
  out[(size_t)row * DD + t]       = (a - mu) * inv * gamma[t] + beta[t];
  out[(size_t)row * DD + t + 256] = (c - mu) * inv * gamma[t + 256] + beta[t + 256];
}

// ---------------------------------------------------------------------------
// Launch
// ---------------------------------------------------------------------------
extern "C" void kernel_launch(void* const* d_in, const int* in_sizes, int n_in,
                              void* d_out, int out_size, void* d_ws,
                              size_t ws_size, hipStream_t stream) {
  const float* query = (const float*)d_in[0];   // [B,L,D]
  const float* key   = (const float*)d_in[1];   // [B,P,D]
  const float* value = (const float*)d_in[2];   // [B,P,D]
  const float* crbf  = (const float*)d_in[3];   // [B,L,P,RBF]
  const int*   kmask = (const int*)d_in[4];     // [B,P]
  const float* Wq = (const float*)d_in[5];
  const float* bq = (const float*)d_in[6];
  const float* Wk = (const float*)d_in[7];
  const float* bk = (const float*)d_in[8];
  const float* Wv = (const float*)d_in[9];
  const float* bv = (const float*)d_in[10];
  const float* Wr = (const float*)d_in[11];
  const float* br = (const float*)d_in[12];
  const float* Wo = (const float*)d_in[13];
  const float* bo = (const float*)d_in[14];
  const float* gamma = (const float*)d_in[15];
  const float* beta  = (const float*)d_in[16];

  float* out_norm = (float*)d_out;                       // [B,L,D]
  float* out_mean = out_norm + (size_t)BB * LL * DD;     // [B,L,P]

  // workspace layout (byte offsets)
  char* w = (char*)d_ws;
  bhalf* qbf    = (bhalf*)(w + 0);                       //  1 MB bf16 [B,L,D]
  bhalf* kbf    = (bhalf*)(w + (1u << 20));              //  4 MB bf16 [B,P,D]
  bhalf* vbf    = (bhalf*)(w + (5u << 20));              //  4 MB bf16 [B,P,D]
  float* logits = (float*)(w + (9u << 20));              // 32 MB f32 [B,H,L,P]
  float* stats  = (float*)(w + (41u << 20));             // 64 KB f32 [B*H*L][2]
  float* attnO  = (float*)(w + (42u << 20));             //  2 MB f32 [B,L,D]
  float* xbuf   = (float*)(w + (44u << 20));             //  2 MB f32 [B,L,D]

  // QKV projections (staged-LDS WMMA GEMMs)
  proj_gemm_bf16<<<(BB * LL / 16) * 4, 256, 0, stream>>>(query, Wq, bq, qbf, BB * LL);
  proj_gemm_bf16<<<(BB * PP / 16) * 4, 256, 0, stream>>>(key, Wk, bk, kbf, BB * PP);
  proj_gemm_bf16<<<(BB * PP / 16) * 4, 256, 0, stream>>>(value, Wv, bv, vbf, BB * PP);

  // logits: B * (L/16) * (P/64) = 1024 blocks, saturates the 210MB rbf stream
  bias_logits<<<BB * 16 * 16, 256, 0, stream>>>(qbf, kbf, crbf, kmask, Wr, br,
                                                logits);
  // softmax stats: 8192 rows / 8 per block
  softmax_stats<<<BB * HH * LL / 8, 256, 0, stream>>>(logits, stats);
  // head-mean output: 1M elements
  attn_mean<<<BB * LL * PP / 256, 256, 0, stream>>>(logits, stats, out_mean);
  // attn @ V: B * (L/16) * H = 512 blocks
  attn_av<<<BB * 16 * HH, 256, 0, stream>>>(vbf, logits, stats, attnO);

  // out-projection + residual, then layernorm
  outproj_residual<<<(BB * LL / 16) * 4, 256, 0, stream>>>(attnO, Wo, bo, query,
                                                           xbuf);
  layernorm_rows<<<BB * LL, 256, 0, stream>>>(xbuf, gamma, beta, out_norm);
}
